// WindowAttentionDecode_6004364280663
// MI455X (gfx1250) — matrix operand order repaired and checked by reading, not compiled, not run on hardware
//
#include <hip/hip_runtime.h>
#include <hip/hip_bf16.h>

typedef __attribute__((ext_vector_type(16))) _Float16 v16h;
typedef __attribute__((ext_vector_type(8)))  _Float16 v8h;
typedef __attribute__((ext_vector_type(4)))  _Float16 v4h;
typedef __attribute__((ext_vector_type(8)))  float    v8f;

#define WMMA_F32_F16(a, b, c) \
  __builtin_amdgcn_wmma_f32_16x16x32_f16(false, (a), false, (b), (short)0, (c), false, false)

__device__ __forceinline__ v16h cat8(v8h lo, v8h hi) {
  return __builtin_shufflevector(lo, hi, 0,1,2,3,4,5,6,7,8,9,10,11,12,13,14,15);
}

// A-matrix fragment (16xK tile, K=32 slice at kbase), ISA 7.12.2:
// lanes 0-15: K {kb+0..7, kb+16..23}; lanes 16-31: K {kb+8..15, kb+24..31}.
__device__ __forceinline__ v16h frag_a_lds(const _Float16* lds, int row, int stride,
                                           int lane, int kbase) {
  int klo = kbase + ((lane & 16) ? 8 : 0);
  v8h lo = *(const v8h*)(lds + row * stride + klo);
  v8h hi = *(const v8h*)(lds + row * stride + klo + 16);
  return cat8(lo, hi);
}

// B-matrix fragment (Kx16 tile); source is N-major rows of K values (B^T
// row-major). lanes 0-15: K {kb+0..7, kb+16..23}; lanes 16-31: K {kb+8..15, kb+24..31}.
__device__ __forceinline__ v16h frag_b_lds(const _Float16* lds, int row, int stride,
                                           int lane, int kbase) {
  int klo = kbase + ((lane & 16) ? 8 : 0);
  v8h lo = *(const v8h*)(lds + row * stride + klo);
  v8h hi = *(const v8h*)(lds + row * stride + klo + 16);
  return cat8(lo, hi);
}

// ---------------------------------------------------------------------------
// Tiled GEMM: C[m][n] = sum_k A[m][k] * W[n][k]  (torch Linear, NT form)
// 128x128x32 block tile, 8 waves (2x4), 4x2 WMMA accumulators per wave.
// Double-buffered LDS; next tile's global loads issued before the WMMAs.
// MODE 0: A=f32 x   -> f16 qbuf[b][h][t][d], (c+bias)*scale
// MODE 1: A=f32 y   -> f16 kbuf/vbuf[b][h][t][d], c+bias
// MODE 2: A=f16 obuf-> f32 d_out[m][n], c+bias
// ---------------------------------------------------------------------------
constexpr int BM = 128, BK = 32, LDK = 40, KDIM = 512, NK = KDIM / BK;

struct StageRegs {
  float4 a[4];
  v8h    ah[2];
  float4 w[4];
};

template <int MODE>
__device__ __forceinline__ void stage_gload(StageRegs& r, const void* Ap,
                                            const float* W, int m0, int n0,
                                            int k0, int tid) {
  if (MODE == 2) {
    const _Float16* Ah = (const _Float16*)Ap;
#pragma unroll
    for (int c = 0; c < 2; ++c) {
      int chunk = tid + 256 * c;          // 512 chunks of 8 halves
      int row = chunk >> 2;
      int col = (chunk & 3) * 8;
      r.ah[c] = *(const v8h*)(Ah + (size_t)(m0 + row) * KDIM + k0 + col);
    }
  } else {
    const float* Af = (const float*)Ap;
#pragma unroll
    for (int c = 0; c < 4; ++c) {
      int chunk = tid + 256 * c;          // 1024 chunks of 4 floats
      int row = chunk >> 3;
      int col = (chunk & 7) * 4;
      r.a[c] = *(const float4*)(Af + (size_t)(m0 + row) * KDIM + k0 + col);
    }
  }
#pragma unroll
  for (int c = 0; c < 4; ++c) {
    int chunk = tid + 256 * c;
    int row = chunk >> 3;
    int col = (chunk & 7) * 4;
    r.w[c] = *(const float4*)(W + (size_t)(n0 + row) * KDIM + k0 + col);
  }
}

template <int MODE>
__device__ __forceinline__ void stage_store(const StageRegs& r, _Float16* As,
                                            _Float16* Bs, int tid) {
  if (MODE == 2) {
#pragma unroll
    for (int c = 0; c < 2; ++c) {
      int chunk = tid + 256 * c;
      int row = chunk >> 2;
      int col = (chunk & 3) * 8;
      *(v8h*)(As + row * LDK + col) = r.ah[c];
    }
  } else {
#pragma unroll
    for (int c = 0; c < 4; ++c) {
      int chunk = tid + 256 * c;
      int row = chunk >> 3;
      int col = (chunk & 7) * 4;
      v4h h;
      h.x = (_Float16)r.a[c].x; h.y = (_Float16)r.a[c].y;
      h.z = (_Float16)r.a[c].z; h.w = (_Float16)r.a[c].w;
      *(v4h*)(As + row * LDK + col) = h;
    }
  }
#pragma unroll
  for (int c = 0; c < 4; ++c) {
    int chunk = tid + 256 * c;
    int row = chunk >> 3;
    int col = (chunk & 7) * 4;
    v4h h;
    h.x = (_Float16)r.w[c].x; h.y = (_Float16)r.w[c].y;
    h.z = (_Float16)r.w[c].z; h.w = (_Float16)r.w[c].w;
    *(v4h*)(Bs + row * LDK + col) = h;
  }
}

template <int MODE>
__global__ __launch_bounds__(256) void gemm_wmma(
    const void* __restrict__ Ap, const float* __restrict__ W,
    const float* __restrict__ bias, void* __restrict__ Out,
    _Float16* __restrict__ kbuf, _Float16* __restrict__ vbuf) {
  __shared__ _Float16 As[2][BM * LDK];
  __shared__ _Float16 Bs[2][BM * LDK];

  const int tid = threadIdx.x;
  const int m0 = blockIdx.x * BM;
  const int n0 = blockIdx.y * BM;
  const int lane = tid & 31;
  const int wave = tid >> 5;
  const int wm = (wave & 1) * 64;   // 2 wave rows of 64
  const int wn = (wave >> 1) * 32;  // 4 wave cols of 32
  const int l15 = lane & 15;

  v8f acc[4][2] = {};

  // prologue: fill buffer 0
  {
    StageRegs sr;
    stage_gload<MODE>(sr, Ap, W, m0, n0, 0, tid);
    stage_store<MODE>(sr, As[0], Bs[0], tid);
  }
  __syncthreads();

#pragma unroll 2
  for (int kt = 0; kt < NK; ++kt) {
    const int cur = kt & 1;
    const bool have_next = (kt + 1 < NK);

    // issue next tile's global loads first (latency overlapped with WMMAs)
    StageRegs sr;
    if (have_next)
      stage_gload<MODE>(sr, Ap, W, m0, n0, (kt + 1) * BK, tid);
    if (kt + 2 < NK) {
      int pr = tid >> 1;
      __builtin_prefetch(W + (size_t)(n0 + pr) * KDIM + (kt + 2) * BK, 0, 0);
    }

    // compute on current buffer
    v16h af[4], bf[2];
#pragma unroll
    for (int mt = 0; mt < 4; ++mt)
      af[mt] = frag_a_lds(As[cur], wm + mt * 16 + l15, LDK, lane, 0);
#pragma unroll
    for (int nt = 0; nt < 2; ++nt)
      bf[nt] = frag_b_lds(Bs[cur], wn + nt * 16 + l15, LDK, lane, 0);
#pragma unroll
    for (int mt = 0; mt < 4; ++mt)
#pragma unroll
      for (int nt = 0; nt < 2; ++nt)
        acc[mt][nt] = WMMA_F32_F16(af[mt], bf[nt], acc[mt][nt]);

    // convert + store next tile into the other buffer while WMMAs drain
    if (have_next)
      stage_store<MODE>(sr, As[cur ^ 1], Bs[cur ^ 1], tid);
    __syncthreads();
  }

  // ---- epilogue ----
  constexpr float QSCALE = 0.17677669529663688f;  // 32^-0.5
#pragma unroll
  for (int mt = 0; mt < 4; ++mt) {
#pragma unroll
    for (int nt = 0; nt < 2; ++nt) {
      const int n = n0 + wn + nt * 16 + l15;
      const float bv = bias[n];
#pragma unroll
      for (int r = 0; r < 8; ++r) {
        const int m = m0 + wm + mt * 16 + ((lane & 16) ? 8 + r : r);
        float c = acc[mt][nt][r] + bv;
        if (MODE == 0) {
          int b = m / 49, t = m % 49;
          int h = n >> 5, d = n & 31;
          ((_Float16*)Out)[(((size_t)b * 16 + h) * 49 + t) * 32 + d] =
              (_Float16)(c * QSCALE);
        } else if (MODE == 1) {
          int b = m / 49, t = m % 49;
          int n2 = n & 511;
          int h = n2 >> 5, d = n2 & 31;
          _Float16* dst = (n < 512) ? kbuf : vbuf;
          dst[(((size_t)b * 16 + h) * 49 + t) * 32 + d] = (_Float16)c;
        } else {
          ((float*)Out)[(size_t)m * 512 + n] = c;
        }
      }
    }
  }
}

// ---------------------------------------------------------------------------
// Per-(b,h) window attention: S = qk^T + (bias+mask as WMMA C), softmax,
// O = P v. 49 tokens padded to 64; hd=32 matches WMMA K exactly. 4 waves,
// each owns a 16-row strip of the 64x64 score matrix.
// ---------------------------------------------------------------------------
__global__ __launch_bounds__(128) void attn_kernel(
    const _Float16* __restrict__ qbuf, const _Float16* __restrict__ kbuf,
    const _Float16* __restrict__ vbuf, const float* __restrict__ mask,
    const float* __restrict__ bias_table, const int* __restrict__ rel_index,
    _Float16* __restrict__ obuf) {
  constexpr int N = 49, HD = 32, H = 16;
  __shared__ _Float16 qs[64 * 40];
  __shared__ _Float16 ks[64 * 40];
  __shared__ _Float16 vt[32 * 72];  // v transposed: [d][token]
  __shared__ float    sb[64 * 65];  // bias+mask -> scores -> exp
  __shared__ _Float16 ps[64 * 72];  // probabilities (f16)

  const int h = blockIdx.x;
  const int b = blockIdx.y;
  const int tid = threadIdx.x;
  const size_t base = ((size_t)b * H + h) * (N * HD);
  const _Float16* qg = qbuf + base;
  const _Float16* kg = kbuf + base;
  const _Float16* vg = vbuf + base;

  for (int idx = tid; idx < 64 * HD; idx += 128) {
    int t = idx >> 5, d = idx & 31;
    _Float16 qv = (_Float16)0.0f, kv = (_Float16)0.0f, vv = (_Float16)0.0f;
    if (t < N) {
      qv = qg[t * HD + d];
      kv = kg[t * HD + d];
      vv = vg[t * HD + d];
    }
    qs[t * 40 + d] = qv;
    ks[t * 40 + d] = kv;
    vt[d * 72 + t] = vv;
  }
  const int w = b & 63;  // window index within image
  for (int idx = tid; idx < 64 * 64; idx += 128) {
    int i = idx >> 6, j = idx & 63;
    float bb = -1e30f;  // padding: drives exp() to 0
    if (i < N && j < N)
      bb = bias_table[rel_index[i * N + j] * H + h] +
           mask[((size_t)w * N + i) * N + j];
    sb[i * 65 + j] = bb;
  }
  __syncthreads();

  const int lane = tid & 31;
  const int wave = tid >> 5;
  const int l15 = lane & 15;
  const int i0 = wave * 16;
  const int rbase = (lane & 16) ? 8 : 0;

  // S = q k^T + bias : bias rides in as the WMMA C operand.
  v16h aq = frag_a_lds(qs, i0 + l15, 40, lane, 0);
#pragma unroll
  for (int nt = 0; nt < 4; ++nt) {
    v16h bk = frag_b_lds(ks, nt * 16 + l15, 40, lane, 0);
    const int j = nt * 16 + l15;
    v8f cb;
#pragma unroll
    for (int r = 0; r < 8; ++r) cb[r] = sb[(i0 + rbase + r) * 65 + j];
    v8f s = WMMA_F32_F16(aq, bk, cb);
#pragma unroll
    for (int r = 0; r < 8; ++r) sb[(i0 + rbase + r) * 65 + j] = s[r];
  }
  __syncthreads();

  // softmax over rows (valid cols exp to >0, padded cols exp to 0)
  if (tid < 64) {
    float mx = -3.0e38f;
    for (int j = 0; j < 64; ++j) mx = fmaxf(mx, sb[tid * 65 + j]);
    float sum = 0.0f;
    for (int j = 0; j < 64; ++j) {
      float e = __expf(sb[tid * 65 + j] - mx);
      sb[tid * 65 + j] = e;
      sum += e;
    }
    float inv = 1.0f / sum;
    for (int j = 0; j < 64; ++j)
      ps[tid * 72 + j] = (_Float16)(sb[tid * 65 + j] * inv);
  }
  __syncthreads();

  // O = P v : K = 64 tokens -> two K=32 WMMA steps; N = 32 dims -> 2 tiles.
#pragma unroll
  for (int dt = 0; dt < 2; ++dt) {
    v8f o = {};
#pragma unroll
    for (int kc = 0; kc < 2; ++kc) {
      v16h ap = frag_a_lds(ps, i0 + l15, 72, lane, kc * 32);
      v16h bv = frag_b_lds(vt, dt * 16 + l15, 72, lane, kc * 32);
      o = WMMA_F32_F16(ap, bv, o);
    }
    const int d = dt * 16 + l15;
#pragma unroll
    for (int r = 0; r < 8; ++r) {
      int t = i0 + rbase + r;
      if (t < N)
        obuf[((size_t)b * N + t) * 512 + h * HD + d] = (_Float16)o[r];
    }
  }
}

extern "C" void kernel_launch(void* const* d_in, const int* in_sizes, int n_in,
                              void* d_out, int out_size, void* d_ws,
                              size_t ws_size, hipStream_t stream) {
  (void)in_sizes; (void)n_in; (void)out_size; (void)ws_size;
  const float* x          = (const float*)d_in[0];
  const float* y          = (const float*)d_in[1];
  const float* mask       = (const float*)d_in[2];
  const float* q_w        = (const float*)d_in[3];
  const float* q_b        = (const float*)d_in[4];
  const float* kv_w       = (const float*)d_in[5];
  const float* kv_b       = (const float*)d_in[6];
  const float* proj_w     = (const float*)d_in[7];
  const float* proj_b     = (const float*)d_in[8];
  const float* bias_table = (const float*)d_in[9];
  const int*   rel_index  = (const int*)d_in[10];

  const size_t QS = (size_t)2048 * 16 * 49 * 32;  // elems of each f16 buffer
  _Float16* qbuf = (_Float16*)d_ws;
  _Float16* kbuf = qbuf + QS;
  _Float16* vbuf = kbuf + QS;
  _Float16* obuf = vbuf + QS;

  const dim3 blk(256);
  // M = 100352 = 784 * 128 exactly; K = 512 = 16 * 32 exactly.
  gemm_wmma<0><<<dim3(784, 4), blk, 0, stream>>>(x, q_w, q_b, qbuf, nullptr, nullptr);
  gemm_wmma<1><<<dim3(784, 8), blk, 0, stream>>>(y, kv_w, kv_b, nullptr, kbuf, vbuf);
  attn_kernel<<<dim3(16, 2048), dim3(128), 0, stream>>>(qbuf, kbuf, vbuf, mask,
                                                        bias_table, rel_index, obuf);
  gemm_wmma<2><<<dim3(784, 4), blk, 0, stream>>>(obuf, proj_w, proj_b, d_out,
                                                 nullptr, nullptr);
}
